// LlamaAttention_28432683499885
// MI455X (gfx1250) — matrix operand order repaired and checked by reading, not compiled, not run on hardware
//
#include <hip/hip_runtime.h>
#include <hip/hip_bf16.h>
#include <math.h>

// Problem constants (T, D, N, K, H) = (2048, 4096, 32, 8, 128)
#define T_SEQ     2048
#define D_MODEL   4096
#define N_HEADS   32
#define KV_HEADS  8
#define HEAD_DIM  128
#define ROPE_THETA 500000.0f

typedef __bf16 bf16_t;
typedef __attribute__((ext_vector_type(16))) __bf16 v16bf;
typedef __attribute__((ext_vector_type(8)))  float  v8f;

__device__ __forceinline__ v8f wmma_bf16(v16bf a, v16bf b, v8f c) {
  // D = A(16x32 bf16) * B(32x16 bf16) + C(16x16 f32)
  return __builtin_amdgcn_wmma_f32_16x16x32_bf16(
      /*neg_a=*/false, a, /*neg_b=*/false, b,
      /*c_mod=*/(short)0, c, /*reuse_a=*/false, /*reuse_b=*/false);
}

// 16-bit A-matrix 16x32 fragment K index (ISA 7.12.2): lanes 0-15 half=0,
// VGPR j -> K = (j>>2)*16 + half*8 + (j&3)*2 (+e)
__device__ __forceinline__ int ka_idx(int half, int j) {
  return ((j >> 2) << 4) | (half << 3) | ((j & 3) << 1);
}
// 16-bit B-matrix 32x16 fragment K index: VGPR j -> K = (j>>2)*16 + half*8 + (j&3)*2 (+e)
__device__ __forceinline__ int kb_idx(int half, int j) {
  return ((j >> 2) << 4) | (half << 3) | ((j & 3) << 1);
}

// ---------------------------------------------------------------------------
// Generic GEMM: C[M x Ntot] = A[M x Kdim] * B[Kdim x Ntot]; fp32 in/out,
// bf16 WMMA compute. Block = 256 threads (8 waves), tile 64(M) x 128(N),
// K-step 32. Waves laid out 4 (M) x 2 (N); each wave owns a 16x64 strip.
// ---------------------------------------------------------------------------
__global__ __launch_bounds__(256)
void gemm_bf16(const float* __restrict__ A, const float* __restrict__ B,
               float* __restrict__ C, int M, int Ntot, int Kdim)
{
  __shared__ bf16_t As[64][40];    // 64x32 tile, +8 pad
  __shared__ bf16_t Bs[32][136];   // 32x128 tile, +8 pad

  const int tid  = threadIdx.x;
  const int lane = tid & 31;
  const int wave = tid >> 5;
  const int wr   = wave & 3;       // wave row -> M offset 16*wr
  const int wc   = wave >> 2;      // wave col -> N offset 64*wc
  const int half = lane >> 4;
  const int l16  = lane & 15;
  const int blockM = blockIdx.x * 64;
  const int blockN = blockIdx.y * 128;

  v8f acc[4];
  for (int i = 0; i < 4; ++i)
    for (int r = 0; r < 8; ++r) acc[i][r] = 0.0f;

  for (int k0 = 0; k0 < Kdim; k0 += 32) {
    // Stage A tile (64x32): 1024 float2 pairs, 4 per thread.
    for (int t = 0; t < 4; ++t) {
      int p   = tid + t * 256;
      int row = p >> 4;
      int col = (p & 15) << 1;
      float2 v = *reinterpret_cast<const float2*>(
          A + (size_t)(blockM + row) * Kdim + k0 + col);
      As[row][col]     = (bf16_t)v.x;
      As[row][col + 1] = (bf16_t)v.y;
    }
    // Stage B tile (32x128): 2048 float2 pairs, 8 per thread.
    for (int t = 0; t < 8; ++t) {
      int p   = tid + t * 256;
      int row = p >> 6;
      int col = (p & 63) << 1;
      float2 v = *reinterpret_cast<const float2*>(
          B + (size_t)(k0 + row) * Ntot + blockN + col);
      Bs[row][col]     = (bf16_t)v.x;
      Bs[row][col + 1] = (bf16_t)v.y;
    }
    if (k0 + 32 < Kdim) {  // warm L2/WGP$ for next K-step (global_prefetch_b8)
      __builtin_prefetch(A + (size_t)(blockM + (tid >> 2)) * Kdim + k0 + 32, 0, 1);
      __builtin_prefetch(B + (size_t)(k0 + 32 + (tid >> 3)) * Ntot + blockN, 0, 1);
    }
    __syncthreads();

    v16bf afrag;
    {
      const int arow = wr * 16 + l16;
      for (int j = 0; j < 8; ++j) {
        int k = ka_idx(half, j);
        afrag[2 * j]     = As[arow][k];
        afrag[2 * j + 1] = As[arow][k + 1];
      }
    }
    for (int nt = 0; nt < 4; ++nt) {
      v16bf bfrag;
      const int bn = wc * 64 + nt * 16 + l16;
      for (int j = 0; j < 8; ++j) {
        int k = kb_idx(half, j);
        bfrag[2 * j]     = Bs[k][bn];
        bfrag[2 * j + 1] = Bs[k + 1][bn];
      }
      acc[nt] = wmma_bf16(afrag, bfrag, acc[nt]);
    }
    __syncthreads();
  }

  // C/D layout: VGPR r -> M = r + 8*half, N = l16.
  for (int nt = 0; nt < 4; ++nt) {
    const int col = blockN + wc * 64 + nt * 16 + l16;
    for (int r = 0; r < 8; ++r) {
      int row = blockM + wr * 16 + r + half * 8;
      C[(size_t)row * Ntot + col] = acc[nt][r];
    }
  }
}

// ---------------------------------------------------------------------------
// RoPE (in-place) on buf[T][heads][H]; rotate-half with theta=5e5.
// ---------------------------------------------------------------------------
__global__ __launch_bounds__(256)
void rope_kernel(float* __restrict__ buf, const int* __restrict__ pos, int heads)
{
  int idx  = blockIdx.x * 256 + threadIdx.x;
  int i    = idx & 63;          // 0..H/2-1
  int rest = idx >> 6;
  int h    = rest % heads;
  int t    = rest / heads;
  if (t >= T_SEQ) return;
  float inv = powf(ROPE_THETA, -(float)i / 64.0f);
  float ang = (float)pos[t] * inv;
  float c = cosf(ang), s = sinf(ang);
  float* p = buf + ((size_t)t * heads + h) * HEAD_DIM;
  float x1 = p[i], x2 = p[i + 64];
  p[i]      = x1 * c - x2 * s;
  p[i + 64] = x2 * c + x1 * s;
}

// ---------------------------------------------------------------------------
// Flash attention: one wave per (16-query block, head). GQA: 4 Q heads share
// each KV head. Online softmax; S and P*V both through bf16 WMMA.
// ---------------------------------------------------------------------------
__global__ __launch_bounds__(32)
void attn_kernel(const float* __restrict__ Q, const float* __restrict__ Kb,
                 const float* __restrict__ Vb, float* __restrict__ ctx)
{
  const int qblk = blockIdx.x;           // 0..T/16-1
  const int head = blockIdx.y;           // 0..N-1
  const int kvh  = head >> 2;            // head / (N/K)
  const int lane = threadIdx.x;
  const int half = lane >> 4;
  const int l16  = lane & 15;
  const int qbase = qblk * 16;
  const float scale = 0.08838834764831845f;  // 1/sqrt(128)

  __shared__ bf16_t Klds[32][136];  // 32 keys x 128 dims (+8 pad)
  __shared__ bf16_t Vlds[32][136];
  __shared__ bf16_t Plds[16][40];   // 16 q x 32 keys (+8 pad)

  // Q as 4 bf16 A-fragments (K chunks of 32 over H=128).
  v16bf aq[4];
  {
    const float* qrow =
        Q + (size_t)(qbase + l16) * (N_HEADS * HEAD_DIM) + head * HEAD_DIM;
    for (int c = 0; c < 4; ++c)
      for (int j = 0; j < 8; ++j) {
        int k = c * 32 + ka_idx(half, j);
        float2 v = *reinterpret_cast<const float2*>(qrow + k);
        aq[c][2 * j]     = (bf16_t)v.x;
        aq[c][2 * j + 1] = (bf16_t)v.y;
      }
  }

  v8f acc[8];                       // 16 q x 128 h ctx accumulator
  for (int ht = 0; ht < 8; ++ht)
    for (int r = 0; r < 8; ++r) acc[ht][r] = 0.0f;
  float m[8], l[8];
  for (int r = 0; r < 8; ++r) { m[r] = -1e30f; l[r] = 0.0f; }

  for (int s0 = 0; s0 < qbase + 16; s0 += 32) {
    // Stage K/V rows (key = s0+lane) as bf16 into LDS.
    {
      const float* krow =
          Kb + (size_t)(s0 + lane) * (KV_HEADS * HEAD_DIM) + kvh * HEAD_DIM;
      const float* vrow =
          Vb + (size_t)(s0 + lane) * (KV_HEADS * HEAD_DIM) + kvh * HEAD_DIM;
      for (int i = 0; i < 32; ++i) {
        float4 kv = *reinterpret_cast<const float4*>(krow + 4 * i);
        float4 vv = *reinterpret_cast<const float4*>(vrow + 4 * i);
        Klds[lane][4 * i + 0] = (bf16_t)kv.x;
        Klds[lane][4 * i + 1] = (bf16_t)kv.y;
        Klds[lane][4 * i + 2] = (bf16_t)kv.z;
        Klds[lane][4 * i + 3] = (bf16_t)kv.w;
        Vlds[lane][4 * i + 0] = (bf16_t)vv.x;
        Vlds[lane][4 * i + 1] = (bf16_t)vv.y;
        Vlds[lane][4 * i + 2] = (bf16_t)vv.z;
        Vlds[lane][4 * i + 3] = (bf16_t)vv.w;
      }
    }
    __syncthreads();

    // S tiles: S[q][key] = sum_h Q[q][h] * K[key][h]  (two 16-key tiles).
    v8f st[2];
    for (int t2 = 0; t2 < 2; ++t2) {
      v8f s;
      for (int r = 0; r < 8; ++r) s[r] = 0.0f;
      for (int c = 0; c < 4; ++c) {
        v16bf bk;
        for (int j = 0; j < 8; ++j) {
          int k = c * 32 + kb_idx(half, j);
          bk[2 * j]     = Klds[t2 * 16 + l16][k];
          bk[2 * j + 1] = Klds[t2 * 16 + l16][k + 1];
        }
        s = wmma_bf16(aq[c], bk, s);
      }
      st[t2] = s;
    }

    // Online softmax per row (rows live per 16-lane half: M = r + 8*half).
    for (int r = 0; r < 8; ++r) {
      int q   = qbase + r + half * 8;
      float v0 = st[0][r] * scale;
      float v1 = st[1][r] * scale;
      v0 = ((s0 + l16)      <= q) ? v0 : -1e30f;
      v1 = ((s0 + 16 + l16) <= q) ? v1 : -1e30f;
      float rm = fmaxf(v0, v1);
      for (int off = 8; off >= 1; off >>= 1)
        rm = fmaxf(rm, __shfl_xor(rm, off, 32));
      float mn    = fmaxf(m[r], rm);
      float alpha = expf(m[r] - mn);
      m[r] = mn;
      float p0 = expf(v0 - mn);
      float p1 = expf(v1 - mn);
      float rs = p0 + p1;
      for (int off = 8; off >= 1; off >>= 1)
        rs += __shfl_xor(rs, off, 32);
      l[r] = l[r] * alpha + rs;
      for (int ht = 0; ht < 8; ++ht) acc[ht][r] *= alpha;
      Plds[r + half * 8][l16]      = (bf16_t)p0;  // transpose P via LDS
      Plds[r + half * 8][16 + l16] = (bf16_t)p1;
    }
    __syncthreads();

    // ctx += P(16x32) * V(32x128): P as A fragment, V tiles as B fragments.
    v16bf ap;
    for (int j = 0; j < 8; ++j) {
      int k = ka_idx(half, j);
      ap[2 * j]     = Plds[l16][k];
      ap[2 * j + 1] = Plds[l16][k + 1];
    }
    for (int ht = 0; ht < 8; ++ht) {
      v16bf bv;
      for (int j = 0; j < 8; ++j) {
        int k = kb_idx(half, j);
        bv[2 * j]     = Vlds[k][ht * 16 + l16];
        bv[2 * j + 1] = Vlds[k + 1][ht * 16 + l16];
      }
      acc[ht] = wmma_bf16(ap, bv, acc[ht]);
    }
    __syncthreads();
  }

  for (int ht = 0; ht < 8; ++ht) {
    int col = head * HEAD_DIM + ht * 16 + l16;
    for (int r = 0; r < 8; ++r) {
      int q = qbase + r + half * 8;
      ctx[(size_t)q * (N_HEADS * HEAD_DIM) + col] = acc[ht][r] / l[r];
    }
  }
}

// ---------------------------------------------------------------------------
extern "C" void kernel_launch(void* const* d_in, const int* in_sizes, int n_in,
                              void* d_out, int out_size, void* d_ws, size_t ws_size,
                              hipStream_t stream)
{
  (void)in_sizes; (void)n_in; (void)out_size; (void)ws_size;

  const float* x   = (const float*)d_in[0];
  const int*   pos = (const int*)  d_in[1];
  const float* Wq  = (const float*)d_in[2];
  const float* Wk  = (const float*)d_in[3];
  const float* Wv  = (const float*)d_in[4];
  const float* Wo  = (const float*)d_in[5];
  float* out = (float*)d_out;

  float* Qb  = (float*)d_ws;                                   // T x N x H
  float* Kb  = Qb + (size_t)T_SEQ * N_HEADS  * HEAD_DIM;       // T x K x H
  float* Vb  = Kb + (size_t)T_SEQ * KV_HEADS * HEAD_DIM;       // T x K x H
  float* Ctx = Vb + (size_t)T_SEQ * KV_HEADS * HEAD_DIM;       // T x N x H

  dim3 blk(256);
  // Q/K/V projections (Wq flat = [D][N*H] row-major, etc.)
  gemm_bf16<<<dim3(T_SEQ / 64, (N_HEADS * HEAD_DIM) / 128), blk, 0, stream>>>(
      x, Wq, Qb, T_SEQ, N_HEADS * HEAD_DIM, D_MODEL);
  gemm_bf16<<<dim3(T_SEQ / 64, (KV_HEADS * HEAD_DIM) / 128), blk, 0, stream>>>(
      x, Wk, Kb, T_SEQ, KV_HEADS * HEAD_DIM, D_MODEL);
  gemm_bf16<<<dim3(T_SEQ / 64, (KV_HEADS * HEAD_DIM) / 128), blk, 0, stream>>>(
      x, Wv, Vb, T_SEQ, KV_HEADS * HEAD_DIM, D_MODEL);

  rope_kernel<<<(T_SEQ * N_HEADS  * 64) / 256, 256, 0, stream>>>(Qb, pos, N_HEADS);
  rope_kernel<<<(T_SEQ * KV_HEADS * 64) / 256, 256, 0, stream>>>(Kb, pos, KV_HEADS);

  attn_kernel<<<dim3(T_SEQ / 16, N_HEADS), dim3(32), 0, stream>>>(Qb, Kb, Vb, Ctx);

  // Output projection: out = Ctx[T x (N*H)] * Wo[(N*H) x D]
  gemm_bf16<<<dim3(T_SEQ / 64, D_MODEL / 128), blk, 0, stream>>>(
      Ctx, Wo, out, T_SEQ, D_MODEL, D_MODEL);
}
